// MultiHeadAttention_35802847380170
// MI455X (gfx1250) — matrix operand order repaired and checked by reading, not compiled, not run on hardware
//
#include <hip/hip_runtime.h>
#include <hip/hip_bf16.h>

// ---------------------------------------------------------------------------
// MultiHeadAttention for MI455X (gfx1250), bf16 WMMA 16x16x32, wave32.
//   1) cvt_w: transpose/convert weights fp32 -> bf16
//   2) proj:  Q/K/V projections (fp32 activations converted in-register)
//   3) attn:  flash-attention, online softmax, LDS P-transpose
//   4) out:   output projection, fp32 result
// All GEMM inner loops are software-pipelined (ping-pong fragment buffers).
// ---------------------------------------------------------------------------

typedef __bf16 bf16;
typedef bf16  bf16x8  __attribute__((ext_vector_type(8)));
typedef bf16  bf16x16 __attribute__((ext_vector_type(16)));
typedef float f32x8   __attribute__((ext_vector_type(8)));

__device__ __forceinline__ bf16 f2bf(float f) { return (bf16)f; }

__device__ __forceinline__ f32x8 wmma_bf16(bf16x16 a, bf16x16 b, f32x8 c) {
  return __builtin_amdgcn_wmma_f32_16x16x32_bf16(
      false, a, false, b, (short)0, c, false, false);
}

// B fragment: b[j] = p[j], 16 contiguous bf16 (32B aligned).
__device__ __forceinline__ bf16x16 ldB(const bf16* p) {
  return *(const bf16x16*)p;
}

// A fragment from two contiguous 8-element runs (16B aligned each).
__device__ __forceinline__ bf16x16 ldA2(const bf16* p0, const bf16* p1) {
  bf16x8 lo = *(const bf16x8*)p0;
  bf16x8 hi = *(const bf16x8*)p1;
  return __builtin_shufflevector(lo, hi, 0,1,2,3,4,5,6,7,8,9,10,11,12,13,14,15);
}

// A fragment from fp32 row: k(j) = kc*32 + hi*16 + j
__device__ __forceinline__ bf16x16 ldA_f32(const float* row, int kc, int hi) {
  bf16x16 a;
  const float* p0 = row + kc * 32 + hi * 16;
  const float* p1 = p0 + 8;
#pragma unroll
  for (int j = 0; j < 8; ++j) { a[j] = f2bf(p0[j]); a[j + 8] = f2bf(p1[j]); }
  return a;
}

// ---------------------------------------------------------------------------
// Stage 1: weight transpose/convert. grid (4096,1,4) x 256 threads.
// z=0..2: W_{q,k,v}[h][d(1024)][e(64)] -> WT[h][e][d] bf16
// z=3:    W_out [h][e(64)][o(1024)]    -> WoT[o][h*64+e] bf16
// ---------------------------------------------------------------------------
__global__ void mha_cvt_w(const float* __restrict__ Wq, const float* __restrict__ Wk,
                          const float* __restrict__ Wv, const float* __restrict__ Wo,
                          bf16* __restrict__ WT) {
  int z = blockIdx.z;
  size_t idx = (size_t)blockIdx.x * 256 + threadIdx.x;   // 0 .. 1048575
  const float* src = (z == 0) ? Wq : (z == 1) ? Wk : (z == 2) ? Wv : Wo;
  bf16* dst = WT + (size_t)z * 1048576;
  float val = src[idx];
  if (z < 3) {
    int e = (int)(idx & 63);
    int d = (int)((idx >> 6) & 1023);
    int h = (int)(idx >> 16);
    dst[((size_t)h * 64 + e) * 1024 + d] = f2bf(val);
  } else {
    int o  = (int)(idx & 1023);
    int he = (int)(idx >> 10);                           // h*64+e
    dst[(size_t)o * 1024 + he] = f2bf(val);
  }
}

// ---------------------------------------------------------------------------
// Stage 2: projections. grid (256, 16, 3) x 32 threads (1 wave).
// which=0: Q, 1: K (stored [h][b*s][64]), 2: V (stored [h][b][e][s]).
// ---------------------------------------------------------------------------
__global__ __launch_bounds__(32) void mha_proj(
    const float* __restrict__ q, const float* __restrict__ k,
    const float* __restrict__ v, const bf16* __restrict__ WT,
    bf16* __restrict__ Qp, bf16* __restrict__ Kp, bf16* __restrict__ Vt) {
  int lane = threadIdx.x;
  int m16 = lane & 15, hi = lane >> 4;
  int tile  = blockIdx.x;   // 0..255  (16-row tiles over b*s = 4096)
  int h     = blockIdx.y;   // 0..15
  int which = blockIdx.z;   // 0,1,2
  const float* src = (which == 0) ? q : (which == 1) ? k : v;
  const bf16* W = WT + (size_t)which * 1048576 + (size_t)h * 65536; // [64 e][1024 d]
  int r0 = tile * 16;
  const float* Arow = src + (size_t)(r0 + m16) * 1024;

  auto loadBf = [&](int kc, bf16x16* dst) {
#pragma unroll
    for (int cb = 0; cb < 4; ++cb)
      dst[cb] = ldB(W + (size_t)(cb * 16 + m16) * 1024 + kc * 32 + hi * 16);
  };

  f32x8 acc[4] = {};
  bf16x16 bF[4], bN[4];
  loadBf(0, bF);
#pragma unroll 2
  for (int kc = 0; kc < 32; kc += 2) {
    loadBf(kc + 1, bN);                       // next chunk in flight
    bf16x16 a = ldA_f32(Arow, kc, hi);
#pragma unroll
    for (int cb = 0; cb < 4; ++cb) acc[cb] = wmma_bf16(a, bF[cb], acc[cb]);
    if (kc + 2 < 32) loadBf(kc + 2, bF);      // chunk after next in flight
    a = ldA_f32(Arow, kc + 1, hi);
#pragma unroll
    for (int cb = 0; cb < 4; ++cb) acc[cb] = wmma_bf16(a, bN[cb], acc[cb]);
  }

  if (which < 2) {
    bf16* out = ((which == 0) ? Qp : Kp) + (size_t)h * 4096 * 64;
#pragma unroll
    for (int cb = 0; cb < 4; ++cb)
#pragma unroll
      for (int i = 0; i < 8; ++i) {
        int m = hi * 8 + i;
        out[(size_t)(r0 + m) * 64 + cb * 16 + m16] = f2bf(acc[cb][i]);
      }
  } else {
    int bb = r0 >> 11;                 // batch
    int s0 = (r0 & 2047) + hi * 8;     // seq start (8 consecutive rows)
#pragma unroll
    for (int cb = 0; cb < 4; ++cb) {
      bf16x8 vv;
#pragma unroll
      for (int i = 0; i < 8; ++i) vv[i] = f2bf(acc[cb][i]);
      *(bf16x8*)(Vt + ((size_t)((h * 2 + bb) * 64 + cb * 16 + m16)) * 2048 + s0) = vv;
    }
  }
}

// ---------------------------------------------------------------------------
// Stage 3: flash attention. grid (128, 2, 16) x 32 threads (1 wave).
// 2-stage pipeline: K frags for block kb+1 + V frags for block kb issue
// before the softmax VALU section (the latency-hiding window).
// ---------------------------------------------------------------------------
__global__ __launch_bounds__(32) void mha_attn(
    const bf16* __restrict__ Qp, const bf16* __restrict__ Kp,
    const bf16* __restrict__ Vt, bf16* __restrict__ Hd) {
  __shared__ __align__(64) bf16 Pl[16 * 32];   // P tile, row-major [m][32 keys]
  int lane = threadIdx.x;
  int m16 = lane & 15, hi = lane >> 4;
  int qt = blockIdx.x;   // 0..127
  int b  = blockIdx.y;   // 0..1
  int h  = blockIdx.z;   // 0..15

  size_t rowbase = (size_t)h * 4096 + (size_t)b * 2048 + (size_t)qt * 16;
  const bf16* Qrow = Qp + (rowbase + m16) * 64;
  bf16x16 a0 = ldA2(Qrow + hi * 16,      Qrow + 8 + hi * 16);   // e 0..31
  bf16x16 a1 = ldA2(Qrow + 32 + hi * 16, Qrow + 40 + hi * 16);  // e 32..63
  const bf16* Kb = Kp + ((size_t)h * 4096 + (size_t)b * 2048) * 64;
  const bf16* Vb = Vt + (size_t)(h * 2 + b) * 64 * 2048;

  f32x8 O[4] = {};
  float mi[8], li[8];
#pragma unroll
  for (int i = 0; i < 8; ++i) { mi[i] = -1e30f; li[i] = 0.0f; }

  const float SCL = 0.125f * 1.44269504088896340736f;  // 1/sqrt(64) * log2(e)

  auto loadK = [&](int key0, bf16x16* kf) {
    const bf16* Kr0 = Kb + (size_t)(key0 + m16) * 64;
    const bf16* Kr1 = Kb + (size_t)(key0 + 16 + m16) * 64;
    kf[0] = ldB(Kr0 + hi * 16);
    kf[1] = ldB(Kr0 + 32 + hi * 16);
    kf[2] = ldB(Kr1 + hi * 16);
    kf[3] = ldB(Kr1 + 32 + hi * 16);
  };

  // One pipelined key-block step: consumes kC (current K frags), refills kN.
  auto step = [&](int kb, bf16x16* kC, bf16x16* kN) {
    int key0 = kb * 32;
    // ---- scores (K frags preloaded last iteration) ----
    f32x8 s[2] = {};
    s[0] = wmma_bf16(a0, kC[0], s[0]);
    s[1] = wmma_bf16(a0, kC[2], s[1]);
    s[0] = wmma_bf16(a1, kC[1], s[0]);
    s[1] = wmma_bf16(a1, kC[3], s[1]);

    // ---- issue this block's V frags + next block's K frags now ----
    bf16x16 vF[4];
#pragma unroll
    for (int cb = 0; cb < 4; ++cb)
      vF[cb] = ldB(Vb + (size_t)(cb * 16 + m16) * 2048 + key0 + hi * 16);
    if (kb < 63) loadK(key0 + 32, kN);
    if (kb < 62)  // speculative prefetch of V two blocks out (global_prefetch)
      __builtin_prefetch(Vb + (size_t)(lane * 2) * 2048 + key0 + 64, 0, 0);

    // ---- online softmax (rows m = hi*8+i; cols across 16-lane half) ----
    float p0v[8], p1v[8], alpha[8];
#pragma unroll
    for (int i = 0; i < 8; ++i) {
      float v0 = s[0][i] * SCL, v1 = s[1][i] * SCL;
      float r = fmaxf(v0, v1);
      r = fmaxf(r, __shfl_xor(r, 1, 32));
      r = fmaxf(r, __shfl_xor(r, 2, 32));
      r = fmaxf(r, __shfl_xor(r, 4, 32));
      r = fmaxf(r, __shfl_xor(r, 8, 32));
      float mnew = fmaxf(mi[i], r);
      float al = __builtin_amdgcn_exp2f(mi[i] - mnew);
      float p0 = __builtin_amdgcn_exp2f(v0 - mnew);
      float p1 = __builtin_amdgcn_exp2f(v1 - mnew);
      float rs = p0 + p1;
      rs += __shfl_xor(rs, 1, 32);
      rs += __shfl_xor(rs, 2, 32);
      rs += __shfl_xor(rs, 4, 32);
      rs += __shfl_xor(rs, 8, 32);
      li[i] = li[i] * al + rs;
      mi[i] = mnew;
      alpha[i] = al;
      p0v[i] = p0; p1v[i] = p1;
    }
#pragma unroll
    for (int cb = 0; cb < 4; ++cb)
#pragma unroll
      for (int i = 0; i < 8; ++i) O[cb][i] *= alpha[i];

    // ---- C-layout -> A-layout transpose of P via LDS ----
    __syncthreads();
#pragma unroll
    for (int i = 0; i < 8; ++i) {
      int m = hi * 8 + i;
      Pl[m * 32 + m16]      = f2bf(p0v[i]);
      Pl[m * 32 + 16 + m16] = f2bf(p1v[i]);
    }
    __syncthreads();
    const bf16* pr = Pl + m16 * 32;
    bf16x16 aP = ldA2(pr + hi * 16, pr + 8 + hi * 16);

    // ---- O += P x V  (K-dim = 32 keys, 4 column blocks of e) ----
#pragma unroll
    for (int cb = 0; cb < 4; ++cb)
      O[cb] = wmma_bf16(aP, vF[cb], O[cb]);
  };

  bf16x16 kA[4], kB4[4];
  loadK(0, kA);
  for (int kb = 0; kb < 64; kb += 2) {
    step(kb,     kA, kB4);   // consumes kA, preloads kB4
    step(kb + 1, kB4, kA);   // consumes kB4, preloads kA
  }

  // ---- normalize and store heads [h][b*s][64] bf16 ----
  float inv[8];
#pragma unroll
  for (int i = 0; i < 8; ++i) inv[i] = 1.0f / li[i];
#pragma unroll
  for (int cb = 0; cb < 4; ++cb)
#pragma unroll
    for (int i = 0; i < 8; ++i) {
      int m = hi * 8 + i;
      Hd[(rowbase + m) * 64 + cb * 16 + m16] = f2bf(O[cb][i] * inv[i]);
    }
}

// ---------------------------------------------------------------------------
// Stage 4: output projection. grid (256, 16) x 32 threads (1 wave).
// out[b*s][1024] = Hd (as [m][k=h*64+e]) x WoT^T ; K chunks of 32.
// ---------------------------------------------------------------------------
__global__ __launch_bounds__(32) void mha_out(
    const bf16* __restrict__ Hd, const bf16* __restrict__ WoT,
    float* __restrict__ out) {
  int lane = threadIdx.x;
  int m16 = lane & 15, hi = lane >> 4;
  int mt = blockIdx.x;   // 0..255
  int nt = blockIdx.y;   // 0..15 (64 output cols each)
  int r0 = mt * 16;

  auto loadBf = [&](int kc, bf16x16* dst) {
#pragma unroll
    for (int cb = 0; cb < 4; ++cb)
      dst[cb] = ldB(WoT + (size_t)(nt * 64 + cb * 16 + m16) * 1024 + kc * 32 + hi * 16);
  };
  auto loadA = [&](int kc) {
    int h  = kc >> 1;
    int e0 = (kc & 1) * 32;
    const bf16* Ar = Hd + ((size_t)h * 4096 + r0 + m16) * 64 + e0;
    return ldA2(Ar + hi * 16, Ar + 8 + hi * 16);
  };

  f32x8 acc[4] = {};
  bf16x16 bF[4], bN[4];
  loadBf(0, bF);
#pragma unroll 2
  for (int kc = 0; kc < 32; kc += 2) {
    loadBf(kc + 1, bN);
    bf16x16 a = loadA(kc);
#pragma unroll
    for (int cb = 0; cb < 4; ++cb) acc[cb] = wmma_bf16(a, bF[cb], acc[cb]);
    if (kc + 2 < 32) loadBf(kc + 2, bF);
    a = loadA(kc + 1);
#pragma unroll
    for (int cb = 0; cb < 4; ++cb) acc[cb] = wmma_bf16(a, bN[cb], acc[cb]);
  }
#pragma unroll
  for (int cb = 0; cb < 4; ++cb)
#pragma unroll
    for (int i = 0; i < 8; ++i) {
      int m = hi * 8 + i;
      out[(size_t)(r0 + m) * 1024 + nt * 64 + cb * 16 + m16] = acc[cb][i];
    }
}

// ---------------------------------------------------------------------------
extern "C" void kernel_launch(void* const* d_in, const int* in_sizes, int n_in,
                              void* d_out, int out_size, void* d_ws, size_t ws_size,
                              hipStream_t stream) {
  const float* q  = (const float*)d_in[0];
  const float* k  = (const float*)d_in[1];
  const float* v  = (const float*)d_in[2];
  const float* Wq = (const float*)d_in[3];
  const float* Wk = (const float*)d_in[4];
  const float* Wv = (const float*)d_in[5];
  const float* Wo = (const float*)d_in[6];
  float* out = (float*)d_out;

  // Workspace layout (bf16 elements), total 40 MiB:
  //   WT @0 (4x1,048,576: WqT,WkT,WvT,WoT), Qp @4,194,304, Kp @8,388,608,
  //   Vt @12,582,912, Hd @16,777,216
  bf16* ws  = (bf16*)d_ws;
  bf16* WT  = ws;
  bf16* WoT = ws + 3145728;
  bf16* Qp  = ws + 4194304;
  bf16* Kp  = ws + 8388608;
  bf16* Vt  = ws + 12582912;
  bf16* Hd  = ws + 16777216;

  mha_cvt_w<<<dim3(4096, 1, 4), 256, 0, stream>>>(Wq, Wk, Wv, Wo, WT);
  mha_proj <<<dim3(256, 16, 3), 32, 0, stream>>>(q, k, v, WT, Qp, Kp, Vt);
  mha_attn <<<dim3(128, 2, 16), 32, 0, stream>>>(Qp, Kp, Vt, Hd);
  mha_out  <<<dim3(256, 16, 1), 32, 0, stream>>>(Hd, WoT, out);
}
